// SpiralConvBlock_33028298506566
// MI455X (gfx1250) — compile-verified, matrix operand order checked by reading, not run on hardware
//
#include <hip/hip_runtime.h>
#include <stdint.h>

#define BB   4
#define LSEQ 2048
#define DIMC 1024
#define DFFC 4096

// LDS-staged GEMM tiling
#define BM   128
#define BN   128
#define BK   32
#define LROW 40     // BK + 8 padding elements -> 80B row stride, conflict-free

typedef __attribute__((ext_vector_type(16))) __bf16    v16bf;
typedef __attribute__((ext_vector_type(8)))  float     v8f;
typedef __attribute__((ext_vector_type(8)))  uint16_t  v8u16;

union Frag16 {
    v16bf b;
    struct { v8u16 lo; v8u16 hi; } h;
};

__device__ __forceinline__ uint16_t f32_to_bf16(float f) {
    uint32_t u = __builtin_bit_cast(uint32_t, f);
    uint32_t r = u + 0x7FFFu + ((u >> 16) & 1u);   // round-to-nearest-even
    return (uint16_t)(r >> 16);
}

// ---------------------------------------------------------------------------
// Transpose fp32 [K][N] -> bf16 [N][K] so WMMA B-fragments are contiguous.
// ---------------------------------------------------------------------------
__global__ void transpose_bf16_kernel(const float* __restrict__ src,
                                      uint16_t* __restrict__ dst,
                                      int K, int N) {
    size_t idx = (size_t)blockIdx.x * blockDim.x + threadIdx.x;
    if (idx >= (size_t)K * N) return;
    int n = (int)(idx / K);
    int k = (int)(idx % K);
    dst[idx] = f32_to_bf16(src[(size_t)k * N + n]);
}

// ---------------------------------------------------------------------------
// LayerNorm over rows of DIMC. mode 0: fp32 out; mode 1: bf16 out.
// ---------------------------------------------------------------------------
#define LN_T 256
__global__ void ln_kernel(const float* __restrict__ x,
                          const float* __restrict__ g,
                          const float* __restrict__ bta,
                          float* __restrict__ outf,
                          uint16_t* __restrict__ outb,
                          int mode) {
    __shared__ float s_sum[LN_T];
    __shared__ float s_sq[LN_T];
    int row = blockIdx.x;
    const float* xr = x + (size_t)row * DIMC;

    float vals[DIMC / LN_T];
    float lsum = 0.f, lsq = 0.f;
#pragma unroll
    for (int i = 0; i < DIMC / LN_T; ++i) {
        float v = xr[threadIdx.x + i * LN_T];
        vals[i] = v;
        lsum += v;
        lsq  += v * v;
    }
    s_sum[threadIdx.x] = lsum;
    s_sq[threadIdx.x]  = lsq;
    __syncthreads();
    for (int off = LN_T / 2; off > 0; off >>= 1) {
        if (threadIdx.x < off) {
            s_sum[threadIdx.x] += s_sum[threadIdx.x + off];
            s_sq[threadIdx.x]  += s_sq[threadIdx.x + off];
        }
        __syncthreads();
    }
    float mean = s_sum[0] * (1.f / DIMC);
    float var  = s_sq[0] * (1.f / DIMC) - mean * mean;
    float inv  = rsqrtf(var + 1e-5f);

#pragma unroll
    for (int i = 0; i < DIMC / LN_T; ++i) {
        int c = threadIdx.x + i * LN_T;
        float v = (vals[i] - mean) * inv * g[c] + bta[c];
        if (mode == 0) outf[(size_t)row * DIMC + c] = v;
        else           outb[(size_t)row * DIMC + c] = f32_to_bf16(v);
    }
}

// ---------------------------------------------------------------------------
// Spiral conv as a first-order complex recurrence + residual:
//   s_t = p * s_{t-1} + p_init * h_t,  s_{-1} = last;  out = Re(s_t) + x_t
// ---------------------------------------------------------------------------
__global__ void scan_kernel(const float* __restrict__ h,
                            const float* __restrict__ x,
                            const float* __restrict__ phazor,
                            const float* __restrict__ phazor_init,
                            const float* __restrict__ last_re,
                            const float* __restrict__ last_im,
                            float* __restrict__ out) {
    int tid = blockIdx.x * blockDim.x + threadIdx.x;   // B*DIMC threads
    int b = tid / DIMC;
    int d = tid % DIMC;

    float pr = phazor[2 * d], pi = phazor[2 * d + 1];
    float pa = sqrtf(pr * pr + pi * pi);
    float sc = __expf(-pa) / pa;                       // p = p/|p| * exp(-|p|)
    pr *= sc; pi *= sc;
    float qr = phazor_init[2 * d], qi = phazor_init[2 * d + 1];
    float sr = last_re[(size_t)b * DIMC + d];
    float si = last_im[(size_t)b * DIMC + d];

    const float* hp = h + (size_t)b * LSEQ * DIMC + d;
    const float* xp = x + (size_t)b * LSEQ * DIMC + d;
    float* op       = out + (size_t)b * LSEQ * DIMC + d;

    for (int t = 0; t < LSEQ; ++t) {
        float hv  = hp[(size_t)t * DIMC];
        float nsr = pr * sr - pi * si + qr * hv;
        float nsi = pr * si + pi * sr + qi * hv;
        sr = nsr; si = nsi;
        op[(size_t)t * DIMC] = sr + xp[(size_t)t * DIMC];
    }
}

// ---------------------------------------------------------------------------
// LDS-staged bf16 WMMA GEMM core.
// Block = 256 threads (8 waves), block tile BM x BN = 128x128, K step 32.
// Wave tile 32x64 (waves arranged 4 x 2) -> acc = 8 v8f (64 VGPRs), no spill.
// Each K step: block stages A(128x32) + B(128x32) bf16 in LDS (16KB global
// traffic for 64 WMMAs = 256B/WMMA), fragments read back via ds_load_b128.
// Global chunk loads are issued before the barrier so their latency overlaps
// the previous step's WMMAs.
// ---------------------------------------------------------------------------
template <int K>
__device__ __forceinline__ void gemm_lds_core(const uint16_t* __restrict__ A,
                                              const uint16_t* __restrict__ Bt,
                                              int Mb, int Nb,
                                              uint16_t* sA, uint16_t* sB,
                                              v8f acc[2][4]) {
    int tid  = threadIdx.x;
    int lane = tid & 31;
    int w    = tid >> 5;
    int mw   = (w & 3) * 32;       // wave M offset in block tile
    int nw   = (w >> 2) * 64;      // wave N offset in block tile
    int r    = lane & 15;
    int kOff = (lane >> 4) << 3;   // 0 or 8

    // staging: 512 8-element chunks per matrix; thread owns chunks tid, tid+256
    int rowS = tid >> 2;           // 0..63
    int qS   = (tid & 3) * 8;      // 0,8,16,24
    const uint16_t* gA0 = A  + (size_t)(Mb + rowS)      * K + qS;
    const uint16_t* gA1 = A  + (size_t)(Mb + rowS + 64) * K + qS;
    const uint16_t* gB0 = Bt + (size_t)(Nb + rowS)      * K + qS;
    const uint16_t* gB1 = Bt + (size_t)(Nb + rowS + 64) * K + qS;
    uint16_t* stA0 = sA + rowS * LROW + qS;
    uint16_t* stA1 = sA + (rowS + 64) * LROW + qS;
    uint16_t* stB0 = sB + rowS * LROW + qS;
    uint16_t* stB1 = sB + (rowS + 64) * LROW + qS;

    const uint16_t* fA = sA + (mw + r) * LROW + kOff;
    const uint16_t* fB = sB + (nw + r) * LROW + kOff;

    for (int kb = 0; kb < K; kb += BK) {
        v8u16 ra0 = *(const v8u16*)(gA0 + kb);
        v8u16 ra1 = *(const v8u16*)(gA1 + kb);
        v8u16 rb0 = *(const v8u16*)(gB0 + kb);
        v8u16 rb1 = *(const v8u16*)(gB1 + kb);
        __syncthreads();                       // previous step's reads done
        *(v8u16*)stA0 = ra0;
        *(v8u16*)stA1 = ra1;
        *(v8u16*)stB0 = rb0;
        *(v8u16*)stB1 = rb1;
        __syncthreads();                       // tiles visible to all waves

        Frag16 af[2], bfr[4];
#pragma unroll
        for (int i = 0; i < 2; ++i) {
            const uint16_t* p = fA + i * 16 * LROW;
            af[i].h.lo = *(const v8u16*)p;
            af[i].h.hi = *(const v8u16*)(p + 16);
        }
#pragma unroll
        for (int j = 0; j < 4; ++j) {
            const uint16_t* p = fB + j * 16 * LROW;
            bfr[j].h.lo = *(const v8u16*)p;
            bfr[j].h.hi = *(const v8u16*)(p + 16);
        }
#pragma unroll
        for (int i = 0; i < 2; ++i)
#pragma unroll
            for (int j = 0; j < 4; ++j)
                acc[i][j] = __builtin_amdgcn_wmma_f32_16x16x32_bf16(
                    false, af[i].b, false, bfr[j].b, (short)0, acc[i][j], false, false);
    }
}

// GEMM1: act = bf16(SiLU(h2 @ w1 + b1)).  A:[8192,1024], Bt:[4096,1024].
__global__ void __launch_bounds__(256) gemm1_kernel(const uint16_t* __restrict__ A,
                                                    const uint16_t* __restrict__ Bt,
                                                    const float* __restrict__ bias,
                                                    uint16_t* __restrict__ act) {
    __shared__ uint16_t sA[BM * LROW];
    __shared__ uint16_t sB[BN * LROW];
    int Mb = blockIdx.x * BM;
    int Nb = blockIdx.y * BN;

    v8f acc[2][4] = {};
    gemm_lds_core<DIMC>(A, Bt, Mb, Nb, sA, sB, acc);

    int lane = threadIdx.x & 31;
    int w    = threadIdx.x >> 5;
    int m0   = Mb + (w & 3) * 32 + ((lane >> 4) << 3);
    int n0   = Nb + (w >> 2) * 64 + (lane & 15);
#pragma unroll
    for (int i = 0; i < 2; ++i)
#pragma unroll
        for (int j = 0; j < 4; ++j) {
            int n = n0 + j * 16;
            float bi = bias[n];
#pragma unroll
            for (int v = 0; v < 8; ++v) {
                int m = m0 + i * 16 + v;
                float val = acc[i][j][v] + bi;
                float sig = 1.f / (1.f + __expf(-val));
                act[(size_t)m * DFFC + n] = f32_to_bf16(val * sig);
            }
        }
}

// GEMM2: out = act @ w2 + b2 + residual (residual pre-staged in out).
__global__ void __launch_bounds__(256) gemm2_kernel(const uint16_t* __restrict__ A,
                                                    const uint16_t* __restrict__ Bt,
                                                    const float* __restrict__ bias,
                                                    float* __restrict__ out) {
    __shared__ uint16_t sA[BM * LROW];
    __shared__ uint16_t sB[BN * LROW];
    int Mb = blockIdx.x * BM;
    int Nb = blockIdx.y * BN;

    v8f acc[2][4] = {};
    gemm_lds_core<DFFC>(A, Bt, Mb, Nb, sA, sB, acc);

    int lane = threadIdx.x & 31;
    int w    = threadIdx.x >> 5;
    int m0   = Mb + (w & 3) * 32 + ((lane >> 4) << 3);
    int n0   = Nb + (w >> 2) * 64 + (lane & 15);
#pragma unroll
    for (int i = 0; i < 2; ++i)
#pragma unroll
        for (int j = 0; j < 4; ++j) {
            int n = n0 + j * 16;
            float bi = bias[n];
#pragma unroll
            for (int v = 0; v < 8; ++v) {
                int m = m0 + i * 16 + v;
                size_t o = (size_t)m * DIMC + n;
                out[o] = acc[i][j][v] + bi + out[o];   // fused residual
            }
        }
}

// ---------------------------------------------------------------------------
// Orchestration. ws layout (bytes):
//   [0,   32M)  h   : fp32 LN1 output              (32MiB)
//   [32M, 48M)  h2  : bf16 LN2 output              (16MiB)
//   [48M, 56M)  w1t : bf16 [DFF][DIM]              (8MiB)
//   [56M, 64M)  w2t : bf16 [DIM][DFF]              (8MiB)
//   [64M, 128M) act : bf16 [B*L][DFF]              (64MiB)
// d_out doubles as x1 (post-conv residual) before GEMM2 overwrites it.
// ---------------------------------------------------------------------------
extern "C" void kernel_launch(void* const* d_in, const int* in_sizes, int n_in,
                              void* d_out, int out_size, void* d_ws, size_t ws_size,
                              hipStream_t stream) {
    const float* x        = (const float*)d_in[0];
    const float* ln_g     = (const float*)d_in[1];
    const float* ln_b     = (const float*)d_in[2];
    const float* phazor   = (const float*)d_in[3];
    const float* ph_init  = (const float*)d_in[4];
    const float* w1       = (const float*)d_in[5];
    const float* b1       = (const float*)d_in[6];
    const float* w2       = (const float*)d_in[7];
    const float* b2       = (const float*)d_in[8];
    const float* last_re  = (const float*)d_in[9];
    const float* last_im  = (const float*)d_in[10];
    float* out            = (float*)d_out;

    char* ws = (char*)d_ws;
    float*    h   = (float*)(ws);
    uint16_t* h2  = (uint16_t*)(ws + (size_t)32 * 1024 * 1024);
    uint16_t* w1t = (uint16_t*)(ws + (size_t)48 * 1024 * 1024);
    uint16_t* w2t = (uint16_t*)(ws + (size_t)56 * 1024 * 1024);
    uint16_t* act = (uint16_t*)(ws + (size_t)64 * 1024 * 1024);

    const int rows = BB * LSEQ;   // 8192

    // Weight convert + transpose (bf16, [N][K] layout for B fragments)
    {
        int total = DIMC * DFFC;
        transpose_bf16_kernel<<<(total + 255) / 256, 256, 0, stream>>>(w1, w1t, DIMC, DFFC);
        transpose_bf16_kernel<<<(total + 255) / 256, 256, 0, stream>>>(w2, w2t, DFFC, DIMC);
    }

    // LN1 -> h (fp32)
    ln_kernel<<<rows, LN_T, 0, stream>>>(x, ln_g, ln_b, h, nullptr, 0);

    // spiral conv recurrence + residual -> out (= x1)
    scan_kernel<<<(BB * DIMC) / 256, 256, 0, stream>>>(h, x, phazor, ph_init,
                                                       last_re, last_im, out);

    // LN2 -> h2 (bf16)
    ln_kernel<<<rows, LN_T, 0, stream>>>(out, ln_g, ln_b, nullptr, h2, 1);

    // FFN GEMM1 + SiLU -> act (bf16)
    gemm1_kernel<<<dim3(rows / BM, DFFC / BN), 256, 0, stream>>>(h2, w1t, b1, act);

    // FFN GEMM2 + bias + residual -> out (fp32)
    gemm2_kernel<<<dim3(rows / BM, DIMC / BN), 256, 0, stream>>>(act, w2t, b2, out);
}
